// MultiHeadAttention_65214783423114
// MI455X (gfx1250) — matrix operand order repaired and checked by reading, not compiled, not run on hardware
//
#include <hip/hip_runtime.h>
#include <hip/hip_bf16.h>
#include <cstdint>
#include <cstddef>

// Problem constants (match reference)
#define BB   2
#define LL   2048
#define DIMD 1024
#define HH   16
#define HD   64
#define MASK_SCALE (-60.0f)   // -10 * 6
#define SM_SCALE   (0.125f)   // 64^-0.5

typedef __bf16 bf16_t;
typedef __bf16 v16bf __attribute__((ext_vector_type(16)));
typedef __bf16 v8bf  __attribute__((ext_vector_type(8)));
typedef float  v8f   __attribute__((ext_vector_type(8)));

union Frag16 { v16bf v; v8bf h[2]; };

// ---- WMMA wrapper: D = A(16x32 bf16) * B(32x16 bf16) + C(16x16 f32) ----
static __device__ inline v8f wmma_bf16(v16bf a, v16bf b, v8f c) {
  return __builtin_amdgcn_wmma_f32_16x16x32_bf16(
      /*neg_a=*/false, a, /*neg_b=*/false, b,
      /*c_mod=*/(short)0, c, /*reuse_a=*/false, /*reuse_b=*/false);
}

// ---- CDNA5 async global->LDS copy (16 bytes per lane), ASYNCcnt-tracked ----
// Builtin signature (from clang diagnostic): param is
// '__attribute__((__vector_size__(16))) int __device__ *' (AS1), LDS side AS3.
typedef int vi4 __attribute__((vector_size(16)));
typedef __attribute__((address_space(1))) vi4* gvi4p;
typedef __attribute__((address_space(3))) vi4* lvi4p;

static __device__ inline void async_copy16(const bf16_t* g, bf16_t* l) {
#if __has_builtin(__builtin_amdgcn_global_load_async_to_lds_b128)
  __builtin_amdgcn_global_load_async_to_lds_b128(
      (gvi4p)(void*)g, (lvi4p)(void*)l, 0, 0);
#else
  uint32_t laddr = (uint32_t)(uintptr_t)(__attribute__((address_space(3))) bf16_t*)l;
  asm volatile("global_load_async_to_lds_b128 %0, %1, off"
               :: "v"(laddr), "v"((uint64_t)(uintptr_t)g)
               : "memory");
#endif
}

static __device__ inline void wait_async0() {
#if __has_builtin(__builtin_amdgcn_s_wait_asynccnt)
  __builtin_amdgcn_s_wait_asynccnt(0);
#else
  asm volatile("s_wait_asynccnt 0x0" ::: "memory");
#endif
}

// A fragment (16x32, bf16): lane holds M-row (lane&15); row32 -> 32 contiguous K.
// elems j = K[kb + j], kb=(lane>>4)*16
static __device__ inline v16bf load_a32(const bf16_t* row32, int lane) {
  const int kb = (lane >> 4) * 16;
  Frag16 f;
  f.h[0] = *(const v8bf*)(row32 + kb);
  f.h[1] = *(const v8bf*)(row32 + kb + 8);
  return f.v;
}

// B fragment (32x16, bf16): lane holds N-column (lane&15); col32 -> 32 contiguous K.
// elems j = K[kb16 + j], kb16=(lane>>4)*16
static __device__ inline v16bf load_b32(const bf16_t* col32, int lane) {
  const int kb = (lane >> 4) * 16;
  Frag16 f;
  f.h[0] = *(const v8bf*)(col32 + kb);
  f.h[1] = *(const v8bf*)(col32 + kb + 8);
  return f.v;
}

// ---------------- conversion kernels ----------------
__global__ void mha_convert_x(const float* __restrict__ x, bf16_t* __restrict__ xb) {
  int idx = blockIdx.x * 256 + threadIdx.x;          // B*L*DIM / 256 blocks
  xb[idx] = (bf16_t)x[idx];
}

// Wt[n][k] = (bf16) W[k][n], three weights selected by blockIdx.z
__global__ void mha_convert_wt(const float* __restrict__ Wq, const float* __restrict__ Wk,
                               const float* __restrict__ Wv, bf16_t* __restrict__ wt) {
  const int z = blockIdx.z;
  const float* W = (z == 0) ? Wq : (z == 1) ? Wk : Wv;
  bf16_t* dst = wt + (size_t)z * DIMD * DIMD;
  int idx = blockIdx.x * 256 + threadIdx.x;          // enumerates [n][k]
  int n = idx >> 10, k = idx & 1023;
  dst[idx] = (bf16_t)W[(size_t)k * DIMD + n];
}

// ---------------- QKV projection GEMM ----------------
// C[m,n] = sum_k xb[m,k] * W[k,n] + bias[n]; W given transposed (Wt[n][k]).
// Block: 256 threads = 8 waves, 128(M) x 64(N) tile; wave w -> M strip w*16.
// Double-buffered async global->LDS staging overlaps DMA with WMMA.
// Output stored bf16 head-split: [b*16+h][l][64]
__global__ void mha_qkv_gemm(const bf16_t* __restrict__ xb, const bf16_t* __restrict__ wt,
                             const float* __restrict__ bq, const float* __restrict__ bk,
                             const float* __restrict__ bv, bf16_t* __restrict__ qkv) {
  const int z = blockIdx.z;
  const bf16_t* W = wt + (size_t)z * DIMD * DIMD;
  const float* bias = (z == 0) ? bq : (z == 1) ? bk : bv;
  bf16_t* outz = qkv + (size_t)z * BB * HH * LL * HD;

  const int tid = threadIdx.x;
  const int lane = tid & 31;
  const int w = tid >> 5;
  const int m0 = blockIdx.x * 128;
  const int n0 = blockIdx.y * 64;

  __shared__ bf16_t As[2][128][32];   // 2 x 8 KB
  __shared__ bf16_t Bts[2][64][32];   // 2 x 4 KB  (Bts[n][k])

  // per-thread staging assignment (async: no VGPR round-trip)
  const int mA = tid >> 1, cA = (tid & 1) * 16;
  const int nB = tid >> 2, cB = (tid & 3) * 8;
  const bf16_t* aSrc = xb + (size_t)(m0 + mA) * DIMD + cA;
  const bf16_t* bSrc = W + (size_t)(n0 + nB) * DIMD + cB;

  auto stage = [&](int buf, int k0) {
    async_copy16(aSrc + k0,     &As[buf][mA][cA]);
    async_copy16(aSrc + k0 + 8, &As[buf][mA][cA + 8]);
    async_copy16(bSrc + k0,     &Bts[buf][nB][cB]);
  };

  v8f acc[4] = {};

  stage(0, 0);
  wait_async0();
  __syncthreads();

  for (int k0 = 0; k0 < DIMD; k0 += 32) {
    const int buf = (k0 >> 5) & 1;
    const bool nxt = (k0 + 32) < DIMD;
    if (nxt) stage(buf ^ 1, k0 + 32);          // DMA next tile during compute

    v16bf af = load_a32(&As[buf][w * 16 + (lane & 15)][0], lane);
#pragma unroll
    for (int nt = 0; nt < 4; ++nt) {
      v16bf bfr = load_b32(&Bts[buf][nt * 16 + (lane & 15)][0], lane);
      acc[nt] = wmma_bf16(af, bfr, acc[nt]);
    }

    if (nxt) wait_async0();
    __syncthreads();
  }

  // epilogue: add bias, split heads, store bf16
#pragma unroll
  for (int nt = 0; nt < 4; ++nt) {
    int ng = n0 + nt * 16 + (lane & 15);
    float bval = bias[ng];
    int hh = ng >> 6, d = ng & 63;
#pragma unroll
    for (int r = 0; r < 8; ++r) {
      int mg = m0 + w * 16 + r + 8 * (lane >> 4);
      int bb = mg >> 11, l = mg & 2047;
      outz[((size_t)(bb * 16 + hh) * LL + l) * HD + d] = (bf16_t)(acc[nt][r] + bval);
    }
  }
}

// ---------------- Flash attention ----------------
// Wave w owns 16 query rows; block = 128 rows. KV tiles of 32, double-buffered:
// K tiles via async global->LDS DMA; V tiles via overlapped load + transposed store.
// S^T = K * Q^T (two 16x16 tiles), online softmax over kv (C-layout M axis),
// O^T accumulated as 4 d-tiles of 16x16: O^T += V^T * P^T.
__global__ void mha_attention(const bf16_t* __restrict__ Q, const bf16_t* __restrict__ K,
                              const bf16_t* __restrict__ V, const float* __restrict__ mask,
                              float* __restrict__ out) {
  const int tid = threadIdx.x;
  const int lane = tid & 31;
  const int w = tid >> 5;
  const int bh = blockIdx.y;                // b*16 + h
  const int b = bh >> 4, h = bh & 15;
  const int qg = blockIdx.x * 128 + w * 16 + (lane & 15);
  const size_t headOff = (size_t)bh * LL * HD;

  // Q strip as two B-fragments (B = Q^T; B-column = Q-row, contiguous d)
  const bf16_t* Qrow = Q + headOff + (size_t)qg * HD;
  const v16bf qf0 = load_b32(Qrow, lane);
  const v16bf qf1 = load_b32(Qrow + 32, lane);

  const float* mrow = mask + ((size_t)b * LL + qg) * LL;

  __shared__ bf16_t Ks[2][32][64];    // K tile,   [kv][d]   (async DMA target)
  __shared__ bf16_t Vts[2][64][32];   // V^T tile, [d][kv]

  // per-thread staging assignment
  const int kvS = tid >> 3, cS = (tid & 7) * 8;
  const bf16_t* kSrc = K + headOff + (size_t)kvS * HD + cS;
  const bf16_t* vSrc = V + headOff + (size_t)kvS * HD + cS;

  auto stageK = [&](int buf, int kv0) {
    async_copy16(kSrc + (size_t)kv0 * HD, &Ks[buf][kvS][cS]);
  };
  auto loadV = [&](int kv0) -> v8bf {
    return *(const v8bf*)(vSrc + (size_t)kv0 * HD);
  };
  auto storeVt = [&](int buf, v8bf vv) {
#pragma unroll
    for (int j = 0; j < 8; ++j) Vts[buf][cS + j][kvS] = vv[j];
  };

  v8f acc[4] = {};
  float mrun = -__builtin_inff();
  float lrun = 0.0f;

  // prologue: stage tile 0
  stageK(0, 0);
  storeVt(0, loadV(0));
  wait_async0();
  __syncthreads();

  for (int kv0 = 0; kv0 < LL; kv0 += 32) {
    const int buf = (kv0 >> 5) & 1;
    const bool nxt = (kv0 + 32) < LL;

    v8bf vnext;
    if (nxt) {                      // overlap next tile's DMA/load with compute
      stageK(buf ^ 1, kv0 + 32);
      vnext = loadV(kv0 + 32);
    }

    // S^T tiles: s0 = kv rows 0-15, s1 = kv rows 16-31 (cols = q)
    v8f s0 = {}, s1 = {};
    const bf16_t* a0 = &Ks[buf][(lane & 15)][0];
    const bf16_t* a1 = &Ks[buf][16 + (lane & 15)][0];
    s0 = wmma_bf16(load_a32(a0, lane),      qf0, s0);
    s0 = wmma_bf16(load_a32(a0 + 32, lane), qf1, s0);
    s1 = wmma_bf16(load_a32(a1, lane),      qf0, s1);
    s1 = wmma_bf16(load_a32(a1 + 32, lane), qf1, s1);

    // scale + mask; kv index for VGPR r: tile0 = kv0 + r + 8*(lane>>4), tile1 = +16
    const int kb = kv0 + 8 * (lane >> 4);
    float st0[8], st1[8];
#pragma unroll
    for (int r = 0; r < 8; ++r) {
      st0[r] = s0[r] * SM_SCALE + mrow[kb + r] * MASK_SCALE;
      st1[r] = s1[r] * SM_SCALE + mrow[kb + 16 + r] * MASK_SCALE;
    }

    // online softmax over kv (own 16 values + partner lane^16)
    float mloc = -__builtin_inff();
#pragma unroll
    for (int r = 0; r < 8; ++r) mloc = fmaxf(mloc, fmaxf(st0[r], st1[r]));
    mloc = fmaxf(mloc, __shfl_xor(mloc, 16, 32));
    const float mnew = fmaxf(mrun, mloc);
    const float corr = __expf(mrun - mnew);
    mrun = mnew;

    float p0[8], p1[8], ssum = 0.0f;
#pragma unroll
    for (int r = 0; r < 8; ++r) {
      p0[r] = __expf(st0[r] - mnew);
      p1[r] = __expf(st1[r] - mnew);
      ssum += p0[r] + p1[r];
    }
    ssum += __shfl_xor(ssum, 16, 32);
    lrun = lrun * corr + ssum;

#pragma unroll
    for (int t = 0; t < 4; ++t)
#pragma unroll
      for (int r = 0; r < 8; ++r) acc[t][r] *= corr;

    // P^T (32 kv x 16 q) as a B-fragment via lane<->lane+16 half-register swap:
    // lane<16 rows {0-15} = {own p0, partner p0}; lane>=16 rows {16-31} = {partner p1, own p1}
    Frag16 pb;
#pragma unroll
    for (int r = 0; r < 8; ++r) {
      float o0 = __shfl_xor(p0[r], 16, 32);
      float o1 = __shfl_xor(p1[r], 16, 32);
      if (lane < 16) { pb.v[r] = (bf16_t)p0[r]; pb.v[r + 8] = (bf16_t)o0; }
      else           { pb.v[r] = (bf16_t)o1;    pb.v[r + 8] = (bf16_t)p1[r]; }
    }

    // O^T += V^T * P^T  (4 d-tiles)
#pragma unroll
    for (int dt = 0; dt < 4; ++dt) {
      v16bf vf = load_a32(&Vts[buf][dt * 16 + (lane & 15)][0], lane);
      acc[dt] = wmma_bf16(vf, pb.v, acc[dt]);
    }

    if (nxt) {
      storeVt(buf ^ 1, vnext);      // transposed V for next tile (other buffer)
      wait_async0();                // K DMA for next tile done (this wave's part)
    }
    __syncthreads();                // publish next tile; retire reads of this one
  }

  // epilogue: normalize and merge heads -> out[b][q][h*64+d], fp32
  const float invl = 1.0f / lrun;
  float* orow = out + ((size_t)b * LL + qg) * DIMD + h * HD;
#pragma unroll
  for (int dt = 0; dt < 4; ++dt)
#pragma unroll
    for (int r = 0; r < 8; ++r)
      orow[dt * 16 + r + 8 * (lane >> 4)] = acc[dt][r] * invl;
}

// ---------------- launch ----------------
extern "C" void kernel_launch(void* const* d_in, const int* in_sizes, int n_in,
                              void* d_out, int out_size, void* d_ws, size_t ws_size,
                              hipStream_t stream) {
  const float* x    = (const float*)d_in[0];
  const float* mask = (const float*)d_in[1];
  const float* Wq   = (const float*)d_in[2];
  const float* bq   = (const float*)d_in[3];
  const float* Wk   = (const float*)d_in[4];
  const float* bk   = (const float*)d_in[5];
  const float* Wv   = (const float*)d_in[6];
  const float* bv   = (const float*)d_in[7];
  float* out = (float*)d_out;

  char* ws = (char*)d_ws;
  const size_t xbBytes  = (size_t)BB * LL * DIMD * sizeof(bf16_t);        // 8 MB
  const size_t wtBytes  = 3ull * DIMD * DIMD * sizeof(bf16_t);            // 6 MB
  const size_t qkvElems = (size_t)BB * HH * LL * HD;                      // 4M per tensor
  bf16_t* xb  = (bf16_t*)ws;
  bf16_t* wt  = (bf16_t*)(ws + xbBytes);
  bf16_t* qkv = (bf16_t*)(ws + xbBytes + wtBytes);                        // 3 x 8 MB

  mha_convert_x<<<dim3((BB * LL * DIMD) / 256), 256, 0, stream>>>(x, xb);
  mha_convert_wt<<<dim3((DIMD * DIMD) / 256, 1, 3), 256, 0, stream>>>(Wq, Wk, Wv, wt);
  mha_qkv_gemm<<<dim3(32, 16, 3), 256, 0, stream>>>(xb, wt, bq, bk, bv, qkv);

  const bf16_t* Qw = qkv;
  const bf16_t* Kw = qkv + qkvElems;
  const bf16_t* Vw = qkv + 2 * qkvElems;
  mha_attention<<<dim3(LL / 128, BB * HH), 256, 0, stream>>>(Qw, Kw, Vw, mask, out);
}
